// CGLayer_43696997269678
// MI455X (gfx1250) — matrix operand order repaired and checked by reading, not compiled, not run on hardware
//
#include <hip/hip_runtime.h>
#include <math.h>

typedef _Float16 half8 __attribute__((ext_vector_type(8)));
typedef _Float16 v16h  __attribute__((ext_vector_type(16)));
typedef float    v8f   __attribute__((ext_vector_type(8)));

#define BATCH 4
#define NN    1024
#define CCH   64
#define COLS  3264   // 192 + 1152 + 1920
#define NPAN  32     // 1024 / 32 k-panels

// ---------------- fragment tables (order matches the python loops) ----------
__device__ const int FR_L1[15]  = {0,1,2, 0,1,1,1,2,2, 0,1,1,2,2,2};
__device__ const int FR_L2[15]  = {0,1,2, 1,0,1,2,1,2, 2,1,2,0,1,2};
__device__ const int FR_LL[15]  = {0,0,0, 1,1,1,1,1,1, 2,2,2,2,2,2};
__device__ const int FR_CG[15]  = {0,1,10, 35,44,53,80,125,170,
                                   245,270,315,390,415,490};   // sequential fill
__device__ const int SX_OFF[3]  = {0,1,4};

// 64-col chunk -> (fragment, k). 51 chunks cover the 3264 columns.
__device__ const int CH_F[51] = {0,1,2,
  3,4,5,6,7,8, 3,4,5,6,7,8, 3,4,5,6,7,8,
  9,10,11,12,13,14, 9,10,11,12,13,14, 9,10,11,12,13,14,
  9,10,11,12,13,14, 9,10,11,12,13,14};
__device__ const int CH_K[51] = {0,0,0,
  0,0,0,0,0,0, 1,1,1,1,1,1, 2,2,2,2,2,2,
  0,0,0,0,0,0, 1,1,1,1,1,1, 2,2,2,2,2,2,
  3,3,3,3,3,3, 4,4,4,4,4,4};

// ---------------- K0: build CG table (615 floats) ---------------------------
static __device__ double d_fact(int n){ double f=1.0; for(int i=2;i<=n;i++) f*=i; return f; }

static __device__ double d_cg(int l1,int m1,int l2,int m2,int L,int M){
  if (m1+m2 != M) return 0.0;
  int dl = l1>l2 ? l1-l2 : l2-l1;
  if (L < dl || L > l1+l2) return 0.0;
  double pre = sqrt((2.0*L+1.0)*d_fact(L+l1-l2)*d_fact(L-l1+l2)*d_fact(l1+l2-L)/d_fact(l1+l2+L+1));
  pre *= sqrt(d_fact(L+M)*d_fact(L-M)*d_fact(l1-m1)*d_fact(l1+m1)*d_fact(l2-m2)*d_fact(l2+m2));
  double s = 0.0;
  for (int k=0; k<=l1+l2-L; k++){
    int d0=k, d1=l1+l2-L-k, d2=l1-m1-k, d3=l2+m2-k, d4=L-l2+m1+k, d5=L-l1-m2+k;
    if (d0<0||d1<0||d2<0||d3<0||d4<0||d5<0) continue;
    double term = (k&1) ? -1.0 : 1.0;
    term /= d_fact(d0)*d_fact(d1)*d_fact(d2)*d_fact(d3)*d_fact(d4)*d_fact(d5);
    s += term;
  }
  return pre*s;
}

__global__ void build_cg_kernel(float* cg){
  if (threadIdx.x==0 && blockIdx.x==0){
    int idx = 0;
    for (int f=0; f<15; f++){
      int l1=FR_L1[f], l2=FR_L2[f], L=FR_LL[f];
      for (int i=0;i<2*l1+1;i++)
        for (int j=0;j<2*l2+1;j++)
          for (int k=0;k<2*L+1;k++)
            cg[idx++] = (float)d_cg(l1, i-l1, l2, j-l2, L, k-L);
    }
  }
}

// ---------------- K1: adjacency int32 -> f16 (0/1 exact) --------------------
__global__ void adj_to_f16(const int* __restrict__ conn, _Float16* __restrict__ adj, int n){
  int g = blockIdx.x*blockDim.x + threadIdx.x;
  if (g < n) adj[g] = (_Float16)(float)conn[g];
}

// ---------------- K2: sx[node][9] = sum_j s_l[b,i,j,m] ----------------------
__global__ void sx_reduce(const float* __restrict__ s0, const float* __restrict__ s1,
                          const float* __restrict__ s2, float* __restrict__ sx){
  int node = blockIdx.x;                       // b*N + i
  int tid  = threadIdx.x;
  float acc[9];
  #pragma unroll
  for (int m=0;m<9;m++) acc[m]=0.f;
  size_t base = (size_t)node * NN;
  for (int j=tid; j<NN; j+=256){
    acc[0] += s0[base + j];
    const float* p1 = s1 + (base + j)*3;
    acc[1]+=p1[0]; acc[2]+=p1[1]; acc[3]+=p1[2];
    const float* p2 = s2 + (base + j)*5;
    acc[4]+=p2[0]; acc[5]+=p2[1]; acc[6]+=p2[2]; acc[7]+=p2[3]; acc[8]+=p2[4];
  }
  __shared__ float red[256];
  for (int m=0;m<9;m++){
    red[tid]=acc[m]; __syncthreads();
    for (int s=128;s>0;s>>=1){ if (tid<s) red[tid]+=red[tid+s]; __syncthreads(); }
    if (tid==0) sx[node*9+m] = red[0];
    __syncthreads();
  }
}

// ---------------- K3: CG product -> K-major panel layout (hi/lo f16) --------
// Pt[((b*32 + panel)*3264 + col)*32 + (m&31)], panel = m>>5.
#define TST 40   // LDS tile stride (halfs)

__global__ __launch_bounds__(256) void cg_product_t(
    const float* __restrict__ v0, const float* __restrict__ v1,
    const float* __restrict__ v2, const float* __restrict__ sx,
    const float* __restrict__ cg,
    _Float16* __restrict__ phi, _Float16* __restrict__ plo){
  int chunk = blockIdx.x;           // 0..50
  int pan   = blockIdx.y;           // 0..31
  int b     = blockIdx.z;           // 0..3
  int tid   = threadIdx.x;

  __shared__ float    cgs[615];
  __shared__ float    sxs[32*9];
  __shared__ _Float16 Th[64*TST];
  __shared__ _Float16 Tl[64*TST];

  for (int t=tid; t<615; t+=256) cgs[t] = cg[t];
  int nodebase = b*NN + pan*32;
  for (int t=tid; t<32*9; t+=256) sxs[t] = sx[(size_t)nodebase*9 + t];
  __syncthreads();

  int f  = CH_F[chunk];
  int k  = CH_K[chunk];
  int l1 = FR_L1[f], l2 = FR_L2[f], L = FR_LL[f];
  int n1 = 2*l1+1, n2 = 2*l2+1, nL = 2*L+1;

  int m  = tid >> 3;                // node within panel, 0..31
  int cb = (tid & 7) * 8;           // 8 channels per thread
  int node = nodebase + m;
  const float* vv = (l1==0) ? (v0 + (size_t)node*1*CCH)
                  : (l1==1) ? (v1 + (size_t)node*3*CCH)
                            : (v2 + (size_t)node*5*CCH);
  const float* sxl = &sxs[m*9 + SX_OFF[l2]];
  const float* cgf = &cgs[FR_CG[f]];

  #pragma unroll
  for (int cs=0; cs<8; cs++){
    int c = cb + cs;
    float x = 0.f;
    for (int i=0;i<n1;i++){
      float vi = vv[i*CCH + c];
      float t  = 0.f;
      for (int j=0;j<n2;j++) t += sxl[j]*cgf[(i*n2+j)*nL + k];
      x += vi * t;
    }
    _Float16 h = (_Float16)x;
    Th[c*TST + m] = h;
    Tl[c*TST + m] = (_Float16)(x - (float)h);
  }
  __syncthreads();

  // write coalesced: thread t -> col cc = t>>2 (0..63), kk = (t&3)*8
  int cc = tid >> 2;
  int kk = (tid & 3) * 8;
  size_t chbase = (((size_t)(b*NPAN + pan))*COLS + chunk*64) * 32;
  *(half8*)(phi + chbase + (size_t)cc*32 + kk) = *(const half8*)&Th[cc*TST + kk];
  *(half8*)(plo + chbase + (size_t)cc*32 + kk) = *(const half8*)&Tl[cc*TST + kk];
}

// ---------------- K4: message passing GEMM via WMMA -------------------------
// out[b] (1024x3264) = adj[b] (1024x1024 f16 0/1) x (phi+plo)[b]
// A double-buffered in LDS; B fragments loaded from global (L2-resident).
// Source order: issue next-A global loads -> hoist all 16 B loads -> 8 wmma
// -> ds_store next-A -> barrier, so loads overlap the wmma chain.
#define BM 128
#define BN 64
#define BK 32
#define AST 40

static __device__ inline v16h ld16(const _Float16* p0, const _Float16* p1){
  half8 x = *(const half8*)p0;
  half8 y = *(const half8*)p1;
  v16h r;
  #pragma unroll
  for (int i=0;i<8;i++){ r[i]=x[i]; r[i+8]=y[i]; }
  return r;
}

__global__ __launch_bounds__(256) void mp_gemm(const _Float16* __restrict__ adj,
                                               const _Float16* __restrict__ phi,
                                               const _Float16* __restrict__ plo,
                                               float* __restrict__ out){
  __shared__ _Float16 Al[2][BM*AST];
  int b   = blockIdx.z;
  int n0  = blockIdx.y * BM;
  int c0  = blockIdx.x * BN;
  int tid = threadIdx.x;
  int w = tid >> 5, lane = tid & 31, lr = lane & 15, g = lane >> 4;
  v8f acc[4] = {v8f{}, v8f{}, v8f{}, v8f{}};
  const _Float16* adjb = adj + (size_t)b*NN*NN;

  // per-thread A staging coords: 32B per thread per K-step
  int srow = tid >> 1;
  int skk  = (tid & 1) * 16;
  const _Float16* asrc = adjb + (size_t)(n0+srow)*NN + skk;

  // prologue: stage k0 = 0 into buffer 0
  {
    const half8* s = (const half8*)asrc;
    *(half8*)&Al[0][srow*AST+skk]   = s[0];
    *(half8*)&Al[0][srow*AST+skk+8] = s[1];
  }
  __syncthreads();

  int arow = w*16 + lr;
  for (int k0=0; k0<NN; k0+=BK){
    int buf = (k0 >> 5) & 1;

    // issue next A-tile global loads early (wait lands before ds_store below)
    half8 na0, na1;
    bool more = (k0 + BK) < NN;
    if (more){
      const half8* s = (const half8*)(asrc + k0 + BK);
      na0 = s[0];
      na1 = s[1];
    }

    // A fragment from current buffer
    v16h a = ld16(&Al[buf][arow*AST + g*16], &Al[buf][arow*AST + g*16 + 8]);

    // hoist all 8 B fragments (16 global_load_b128) ahead of the wmma chain
    size_t pbase = (((size_t)(b*NPAN + (k0 >> 5)))*COLS + c0) * 32;
    v16h bh[4], bl[4];
    #pragma unroll
    for (int t=0;t<4;t++){
      size_t off = pbase + (size_t)(t*16 + lr)*32 + g*16;
      bh[t] = ld16(phi + off, phi + off + 8);
      bl[t] = ld16(plo + off, plo + off + 8);
    }
    #pragma unroll
    for (int t=0;t<4;t++){
      acc[t] = __builtin_amdgcn_wmma_f32_16x16x32_f16(false, a, false, bh[t],
                                                      (short)0, acc[t], false, false);
      acc[t] = __builtin_amdgcn_wmma_f32_16x16x32_f16(false, a, false, bl[t],
                                                      (short)0, acc[t], false, false);
    }

    // store next A tile into the other buffer, then one barrier per K-step
    if (more){
      *(half8*)&Al[buf^1][srow*AST+skk]   = na0;
      *(half8*)&Al[buf^1][srow*AST+skk+8] = na1;
    }
    __syncthreads();
  }

  // D layout: VGPR r -> (M = r + 8*(lane/16), N = lane%16)
  #pragma unroll
  for (int t=0;t<4;t++){
    int col = c0 + t*16 + lr;
    #pragma unroll
    for (int r=0;r<8;r++){
      int row = n0 + w*16 + r + 8*g;
      out[((size_t)b*NN + row)*COLS + col] = acc[t][r];
    }
  }
}

// ---------------- K5..K7: per-L Frobenius norm + scale (deterministic) ------
__global__ void sumsq_kernel(const float* __restrict__ out, float* __restrict__ partials, int total){
  int tid = threadIdx.x;
  float a0=0.f, a1=0.f, a2=0.f;
  for (int g = blockIdx.x*256 + tid; g < total; g += 1024*256){
    float x = out[g];
    int col = g % COLS;
    float xx = x*x;
    if (col < 192) a0 += xx; else if (col < 1344) a1 += xx; else a2 += xx;
  }
  __shared__ float red[256];
  float a[3] = {a0,a1,a2};
  for (int L=0;L<3;L++){
    red[tid]=a[L]; __syncthreads();
    for (int s=128;s>0;s>>=1){ if (tid<s) red[tid]+=red[tid+s]; __syncthreads(); }
    if (tid==0) partials[blockIdx.x*3+L] = red[0];
    __syncthreads();
  }
}

__global__ void scale_compute(const float* __restrict__ partials, float* __restrict__ scales){
  int tid = threadIdx.x;
  float a[3] = {0.f,0.f,0.f};
  for (int i=tid; i<1024; i+=256){
    a[0]+=partials[i*3+0]; a[1]+=partials[i*3+1]; a[2]+=partials[i*3+2];
  }
  __shared__ float red[256];
  for (int L=0;L<3;L++){
    red[tid]=a[L]; __syncthreads();
    for (int s=128;s>0;s>>=1){ if (tid<s) red[tid]+=red[tid+s]; __syncthreads(); }
    if (tid==0) scales[L] = 64.0f / ((float)(2*L+1) * sqrtf(red[0]));
    __syncthreads();
  }
}

__global__ void apply_scale(float* __restrict__ out, const float* __restrict__ scales, int total){
  int g = blockIdx.x*blockDim.x + threadIdx.x;
  if (g >= total) return;
  int col = g % COLS;
  int L = (col < 192) ? 0 : ((col < 1344) ? 1 : 2);
  out[g] *= scales[L];
}

// ---------------- launch ----------------------------------------------------
extern "C" void kernel_launch(void* const* d_in, const int* in_sizes, int n_in,
                              void* d_out, int out_size, void* d_ws, size_t ws_size,
                              hipStream_t stream) {
  (void)in_sizes; (void)n_in;
  const float* v0 = (const float*)d_in[0];
  const float* v1 = (const float*)d_in[1];
  const float* v2 = (const float*)d_in[2];
  const float* s0 = (const float*)d_in[3];
  const float* s1 = (const float*)d_in[4];
  const float* s2 = (const float*)d_in[5];
  const int*   cn = (const int*)  d_in[6];
  float* outp = (float*)d_out;

  // workspace layout (bytes)
  const size_t OFF_CG    = 0;                          // 615 f32
  const size_t OFF_SCALE = 4096;                       // 3 f32
  const size_t OFF_PART  = 4352;                       // 1024*3 f32
  const size_t OFF_SX    = 16640;                      // B*N*9 f32 = 147456
  const size_t OFF_ADJ   = 164096;                     // B*N*N f16 = 8 MiB
  const size_t OFF_PHI   = OFF_ADJ + (size_t)BATCH*NN*NN*2;        // 8552704
  const size_t OFF_PLO   = OFF_PHI + (size_t)BATCH*NN*COLS*2;      // 35291392
  const size_t NEED      = OFF_PLO + (size_t)BATCH*NN*COLS*2;      // ~62 MB
  if (ws_size < NEED) return;

  char* ws = (char*)d_ws;
  float*     cg     = (float*)(ws + OFF_CG);
  float*     scales = (float*)(ws + OFF_SCALE);
  float*     part   = (float*)(ws + OFF_PART);
  float*     sx     = (float*)(ws + OFF_SX);
  _Float16*  adj    = (_Float16*)(ws + OFF_ADJ);
  _Float16*  phi    = (_Float16*)(ws + OFF_PHI);     // K-major panel layout
  _Float16*  plo    = (_Float16*)(ws + OFF_PLO);

  const int nadj  = BATCH*NN*NN;
  const int total = out_size;                          // = BATCH*NN*COLS

  build_cg_kernel<<<1, 64, 0, stream>>>(cg);
  adj_to_f16<<<(nadj+255)/256, 256, 0, stream>>>(cn, adj, nadj);
  sx_reduce<<<BATCH*NN, 256, 0, stream>>>(s0, s1, s2, sx);
  cg_product_t<<<dim3(51, NPAN, BATCH), 256, 0, stream>>>(v0, v1, v2, sx, cg, phi, plo);
  mp_gemm<<<dim3(COLS/BN, NN/BM, BATCH), 256, 0, stream>>>(adj, phi, plo, outp);
  sumsq_kernel<<<1024, 256, 0, stream>>>(outp, part, total);
  scale_compute<<<1, 256, 0, stream>>>(part, scales);
  apply_scale<<<(total+255)/256, 256, 0, stream>>>(outp, scales, total);
}
